// ScannedRNN_73426760892768
// MI455X (gfx1250) — compile-verified, matrix operand order checked by reading, not run on hardware
//
#include <hip/hip_runtime.h>

// ---------------- problem constants ----------------
#define L_ 512
#define B_ 128
#define D_ 256
#define H_ 512
#define K_ (D_ + H_)            // 768 combined K ([x | h])
#define N4H (4 * H_)            // 2048
#define NB 32                   // persistent blocks (one 16-wide h-col tile each)
#define LROW 776                // padded LDS row stride (768 + 8): kills ds bank conflicts

// ---------------- types ----------------
typedef __bf16 bf16;
typedef __attribute__((ext_vector_type(16))) __bf16 v16bf;
typedef __attribute__((ext_vector_type(8)))  float  v8f;
typedef __attribute__((ext_vector_type(4)))  unsigned int v4u;

union Frag { v16bf v; v4u u[2]; };

// ws layout (bytes)
#define OFF_CNT 0
#define OFF_HB  128                                   // bf16 h buffer (B x H)
#define OFF_WT  (OFF_HB + B_ * H_ * 2)                // bf16 Wt (4H x K), row-major N x K
#define OFF_XB  (OFF_WT + N4H * K_ * 2)               // bf16 x (L x B x D)
// total: ~36.9 MB

__device__ __forceinline__ bf16 f2bf(float f) {
  unsigned int u = __builtin_bit_cast(unsigned int, f);
  u += 0x7FFFu + ((u >> 16) & 1u);                    // round to nearest even
  unsigned short s = (unsigned short)(u >> 16);
  return __builtin_bit_cast(bf16, s);
}

__device__ __forceinline__ float sigm(float x) { return 1.0f / (1.0f + __expf(-x)); }
__device__ __forceinline__ float tanh_(float x) {
  float e = __expf(-2.0f * x);
  return (1.0f - e) / (1.0f + e);
}

// ---------------- setup: bf16 conversions + Wt transpose + barrier reset ----------------
__global__ void lstm_setup_kernel(const float* __restrict__ x,
                                  const float* __restrict__ init_h,
                                  const float* __restrict__ Wi,
                                  const float* __restrict__ Wh,
                                  unsigned char* __restrict__ ws) {
  long i = (long)blockIdx.x * blockDim.x + threadIdx.x;
  unsigned int* cnt = (unsigned int*)(ws + OFF_CNT);
  bf16* hb = (bf16*)(ws + OFF_HB);
  bf16* Wt = (bf16*)(ws + OFF_WT);
  bf16* xb = (bf16*)(ws + OFF_XB);

  if (i == 0) *cnt = 0u;
  if (i < (long)B_ * H_) hb[i] = f2bf(init_h[i]);
  if (i < (long)N4H * K_) {
    int n = (int)(i / K_);
    int k = (int)(i % K_);
    float w = (k < D_) ? Wi[(long)k * N4H + n] : Wh[(long)(k - D_) * N4H + n];
    Wt[i] = f2bf(w);
  }
  if (i < (long)L_ * B_ * D_) xb[i] = f2bf(x[i]);
}

// ---------------- persistent scanned-LSTM kernel ----------------
__global__ void __launch_bounds__(256)
lstm_scan_kernel(const unsigned char* __restrict__ resets,
                 const float* __restrict__ init_c,
                 const float* __restrict__ bias_in,
                 float* __restrict__ out,
                 unsigned char* __restrict__ ws) {
  unsigned int* cnt = (unsigned int*)(ws + OFF_CNT);
  bf16* hb       = (bf16*)(ws + OFF_HB);
  const bf16* Wt = (const bf16*)(ws + OFF_WT);
  const bf16* xb = (const bf16*)(ws + OFF_XB);

  // Time-invariant weight fragments live in LDS (staged once, read all 512 steps).
  // 4 gates x 16 rows x LROW bf16 = 99,328 B  (well under 320 KB/WGP on CDNA5).
  __shared__ __align__(16) bf16 lwt[4 * 16 * LROW];

  const int lane = threadIdx.x & 31;
  const int wv   = threadIdx.x >> 5;    // wave id 0..7 -> batch rows [16wv,16wv+16)
  const int l15  = lane & 15;
  const int lhf  = lane >> 4;           // 0/1: which K half-group (ISA 16-bit A layout)
  const int hc0  = blockIdx.x * 16;     // this block's 16 h-columns
  const int mrowA = wv * 16 + l15;      // A-fragment row for this lane
  const int col   = hc0 + l15;          // C/D column for this lane

  // ---- one-time stage of this block's Wt stripe into LDS ----
  for (int c = threadIdx.x; c < 4 * 16 * (K_ / 8); c += 256) {
    int row = c / (K_ / 8);             // 0..63 = g*16 + n
    int off = (c % (K_ / 8)) * 8;       // element offset within row, step 8
    int g = row >> 4, n = row & 15;
    const bf16* src = Wt + (long)(g * H_ + hc0 + n) * K_ + off;
    *(v4u*)(lwt + row * LROW + off) = *(const v4u*)src;
  }
  __syncthreads();

  // per-gate bias (C/D layout: lane holds column l15 for all 8 acc rows)
  float bias[4];
#pragma unroll
  for (int g = 0; g < 4; ++g) bias[g] = bias_in[g * H_ + col];

  // per-lane LDS B-fragment row bases
  const bf16* lrow[4];
#pragma unroll
  for (int g = 0; g < 4; ++g) lrow[g] = lwt + (g * 16 + l15) * LROW;

  // cell state tile lives in registers for the whole scan:
  // lane holds c[row = 16wv + r + 8*lhf][col] for r=0..7
  float creg[8];
#pragma unroll
  for (int r = 0; r < 8; ++r) {
    int row = wv * 16 + r + 8 * lhf;
    creg[r] = init_c[(long)row * H_ + col];
  }

  const v4u vzero = {0u, 0u, 0u, 0u};

  for (int t = 0; t < L_; ++t) {
    const unsigned char* rs = resets + (long)t * B_;

    // reset-mask the resident cell state
#pragma unroll
    for (int r = 0; r < 8; ++r) {
      int row = wv * 16 + r + 8 * lhf;
      if (rs[row]) creg[r] = 0.0f;
    }
    const bool rstA = rs[mrowA] != 0;   // masks h-contribution rows of A

    // init accumulators with bias
    v8f acc[4];
#pragma unroll
    for (int g = 0; g < 4; ++g)
#pragma unroll
      for (int e = 0; e < 8; ++e) acc[g][e] = bias[g];

    const bf16* arow_x = xb + ((long)t * B_ + mrowA) * D_;
    const bf16* arow_h = hb + (long)mrowA * H_;

    // --- K chunks 0..7: x part (never masked); A from global, B from LDS ---
#pragma unroll 2
    for (int kc = 0; kc < D_ / 32; ++kc) {
      Frag af;
      const bf16* p = arow_x + kc * 32;
      af.u[0] = *(const v4u*)(p + lhf * 8);
      af.u[1] = *(const v4u*)(p + 16 + lhf * 8);
#pragma unroll
      for (int g = 0; g < 4; ++g) {
        Frag bfr;
        const bf16* q = lrow[g] + kc * 32;
        bfr.u[0] = *(const v4u*)(q + lhf * 8);
        bfr.u[1] = *(const v4u*)(q + 16 + lhf * 8);
        acc[g] = __builtin_amdgcn_wmma_f32_16x16x32_bf16(
            false, af.v, false, bfr.v, (short)0, acc[g], false, false);
      }
    }

    // --- K chunks 8..23: h part (zeroed for reset rows via cndmask; EXEC stays full) ---
#pragma unroll 2
    for (int kc = 0; kc < H_ / 32; ++kc) {
      Frag af;
      const bf16* p = arow_h + kc * 32;
      af.u[0] = *(const v4u*)(p + lhf * 8);
      af.u[1] = *(const v4u*)(p + 16 + lhf * 8);
      if (rstA) { af.u[0] = vzero; af.u[1] = vzero; }
#pragma unroll
      for (int g = 0; g < 4; ++g) {
        Frag bfr;
        const bf16* q = lrow[g] + (D_ + kc * 32);
        bfr.u[0] = *(const v4u*)(q + lhf * 8);
        bfr.u[1] = *(const v4u*)(q + 16 + lhf * 8);
        acc[g] = __builtin_amdgcn_wmma_f32_16x16x32_bf16(
            false, af.v, false, bfr.v, (short)0, acc[g], false, false);
      }
    }

    // --- elementwise LSTM math + state/output writes ---
    float* orow = out + (long)t * B_ * H_;
#pragma unroll
    for (int r = 0; r < 8; ++r) {
      float iv = sigm(acc[0][r]);
      float fv = sigm(acc[1][r]);
      float gv = tanh_(acc[2][r]);
      float ov = sigm(acc[3][r]);
      float cn = fv * creg[r] + iv * gv;
      creg[r] = cn;
      float hn = ov * tanh_(cn);
      int row = wv * 16 + r + 8 * lhf;
      orow[(long)row * H_ + col] = hn;          // ys[t]
      hb[(long)row * H_ + col]  = f2bf(hn);     // recurrent state for t+1
    }

    // prefetch next step's x row while we sit in the grid barrier
    if (t + 1 < L_) {
      __builtin_prefetch(xb + ((long)(t + 1) * B_ + mrowA) * D_ + (long)lhf * 8, 0, 0);
    }

    // --- grid-wide barrier (32 blocks, monotonic counter) ---
    __threadfence();
    __syncthreads();
    if (threadIdx.x == 0) {
      atomicAdd(cnt, 1u);
      const unsigned int target = (unsigned int)(NB * (t + 1));
      while (atomicAdd(cnt, 0u) < target) { __builtin_amdgcn_s_sleep(2); }
    }
    __syncthreads();
  }
}

// ---------------- launcher ----------------
extern "C" void kernel_launch(void* const* d_in, const int* in_sizes, int n_in,
                              void* d_out, int out_size, void* d_ws, size_t ws_size,
                              hipStream_t stream) {
  const float*         x      = (const float*)d_in[0];
  const unsigned char* resets = (const unsigned char*)d_in[1];  // jax bool -> 1 byte
  const float*         init_c = (const float*)d_in[2];
  // d_in[3] = init_h (consumed by setup kernel)
  const float*         Wi     = (const float*)d_in[4];
  const float*         Wh     = (const float*)d_in[5];
  const float*         b      = (const float*)d_in[6];
  unsigned char*       ws     = (unsigned char*)d_ws;

  const long ntot = (long)L_ * B_ * D_;                 // dominant setup range
  const int  thr  = 256;
  const unsigned grid = (unsigned)((ntot + thr - 1) / thr);

  lstm_setup_kernel<<<dim3(grid), dim3(thr), 0, stream>>>(
      x, (const float*)d_in[3], Wi, Wh, ws);

  lstm_scan_kernel<<<dim3(NB), dim3(256), 0, stream>>>(
      resets, init_c, b, (float*)d_out, ws);
}
